// Sup_ContrastiveLearning_14585708937233
// MI455X (gfx1250) — matrix operand order repaired and checked by reading, not compiled, not run on hardware
//
#include <hip/hip_runtime.h>
#include <hip/hip_bf16.h>

typedef __attribute__((ext_vector_type(16))) __bf16 v16bf;
typedef __attribute__((ext_vector_type(8)))  __bf16 v8bf;
typedef __attribute__((ext_vector_type(8)))  float  v8f;

#define N_ROWS 8192
#define HDIM   1024
#define TINV   10.0f   // 1 / temperature

// ---------------------------------------------------------------------------
// Phase 1: row-normalize, split into bf16 hi/lo pair (bf16x3 trick -> ~fp32
// accuracy through the WMMA path).
// ---------------------------------------------------------------------------
__global__ void __launch_bounds__(256)
normalize_split_kernel(const float* __restrict__ emb,
                       __bf16* __restrict__ xhi,
                       __bf16* __restrict__ xlo) {
    __shared__ float red[256];
    const int row = blockIdx.x;
    const int t   = threadIdx.x;
    const float* rp = emb + (size_t)row * HDIM;

    float vals[4];
    float ss = 0.f;
#pragma unroll
    for (int i = 0; i < 4; ++i) {
        vals[i] = rp[t + 256 * i];
        ss += vals[i] * vals[i];
    }
    red[t] = ss;
    __syncthreads();
    for (int s = 128; s > 0; s >>= 1) {
        if (t < s) red[t] += red[t + s];
        __syncthreads();
    }
    const float inv = 1.0f / fmaxf(sqrtf(red[0]), 1e-12f);
#pragma unroll
    for (int i = 0; i < 4; ++i) {
        float xn = vals[i] * inv;
        __bf16 h = (__bf16)xn;
        __bf16 l = (__bf16)(xn - (float)h);
        size_t idx = (size_t)row * HDIM + t + 256 * i;
        xhi[idx] = h;
        xlo[idx] = l;
    }
}

// ---------------------------------------------------------------------------
// Zero the accumulator arrays (harness poisons workspace).
// ---------------------------------------------------------------------------
__global__ void zero_kernel(float* __restrict__ p, int n) {
    int i = blockIdx.x * blockDim.x + threadIdx.x;
    if (i < n) p[i] = 0.f;
}

// ---------------------------------------------------------------------------
// Phase 2: one wave per (16-row M-tile, 128-col chunk). Accumulate 8 C tiles
// over K=1024 with v_wmma_f32_16x16x32_bf16 (3 WMMAs/k-step for the hi/lo
// split), then fold exp(10*sim) into per-row total & same-label sums.
//
// A frag (16x32 bf16, row-major x): lane m = lane&15 is row m; K values are
// b..b+15, b = 16*(lane>>4)  -> two contiguous 16B loads.
// B frag (32x16 bf16 = x^T tile): lane n = lane&15 is column n = a ROW of x;
// K values are base..base+15, base = 16*(lane>>4) -> one contiguous 32B load.
// ---------------------------------------------------------------------------
__global__ void __launch_bounds__(32)
sim_wmma_kernel(const __bf16* __restrict__ xhi,
                const __bf16* __restrict__ xlo,
                const int*    __restrict__ labels,
                float* __restrict__ tot,
                float* __restrict__ pos) {
    const int lane = threadIdx.x;
    const int half = lane >> 4;   // 0 or 1
    const int l15  = lane & 15;
    const int row0 = blockIdx.x * 16;
    const int col0 = blockIdx.y * 128;

    v8f c[8] = {};

    const __bf16* aHiP = xhi + (size_t)(row0 + l15) * HDIM + half * 16;
    const __bf16* aLoP = xlo + (size_t)(row0 + l15) * HDIM + half * 16;

    for (int kk = 0; kk < HDIM; kk += 32) {
        v8bf a0h = *(const v8bf*)(aHiP + kk);
        v8bf a1h = *(const v8bf*)(aHiP + kk + 8);
        v8bf a0l = *(const v8bf*)(aLoP + kk);
        v8bf a1l = *(const v8bf*)(aLoP + kk + 8);
        v16bf aHi, aLo;
#pragma unroll
        for (int i = 0; i < 8; ++i) {
            aHi[i]     = a0h[i];
            aHi[8 + i] = a1h[i];
            aLo[i]     = a0l[i];
            aLo[8 + i] = a1l[i];
        }
#pragma unroll
        for (int nt = 0; nt < 8; ++nt) {
            size_t boff = (size_t)(col0 + nt * 16 + l15) * HDIM + kk + half * 16;
            v16bf bHi = *(const v16bf*)(xhi + boff);
            v16bf bLo = *(const v16bf*)(xlo + boff);
            c[nt] = __builtin_amdgcn_wmma_f32_16x16x32_bf16(
                        false, aHi, false, bHi, (short)0, c[nt], false, false);
            c[nt] = __builtin_amdgcn_wmma_f32_16x16x32_bf16(
                        false, aHi, false, bLo, (short)0, c[nt], false, false);
            c[nt] = __builtin_amdgcn_wmma_f32_16x16x32_bf16(
                        false, aLo, false, bHi, (short)0, c[nt], false, false);
        }
    }

    // Epilogue: exp + masked accumulation.
    // C layout: lanes 0-15 VGPR v -> (M=v, N=lane); lanes 16-31 -> (M=v+8, N=lane-16).
    int rlab[8];
#pragma unroll
    for (int v = 0; v < 8; ++v) rlab[v] = labels[row0 + v + 8 * half];

    float rtot[8] = {};
    float rpos[8] = {};
#pragma unroll
    for (int nt = 0; nt < 8; ++nt) {
        const int gcol = col0 + nt * 16 + l15;
        const int clab = labels[gcol];
#pragma unroll
        for (int v = 0; v < 8; ++v) {
            const int grow = row0 + v + 8 * half;
            float e = __expf(c[nt][v] * TINV);
            bool self = (grow == gcol);
            rtot[v] += self ? 0.f : e;
            rpos[v] += (!self && (rlab[v] == clab)) ? e : 0.f;
        }
    }

    // Reduce the 16 columns held across each half-wave, then atomically fold
    // the partial (this wave covers 128 of 8192 columns) into global sums.
#pragma unroll
    for (int v = 0; v < 8; ++v) {
        float tsum = rtot[v], psum = rpos[v];
#pragma unroll
        for (int m = 8; m >= 1; m >>= 1) {
            tsum += __shfl_xor(tsum, m, 32);
            psum += __shfl_xor(psum, m, 32);
        }
        if (l15 == 0) {
            atomicAdd(&tot[row0 + v + 8 * half], tsum);
            atomicAdd(&pos[row0 + v + 8 * half], psum);
        }
    }
}

// ---------------------------------------------------------------------------
// Phase 3: label histogram -> pos_cnt; per-sample loss; masked mean.
// ---------------------------------------------------------------------------
__global__ void __launch_bounds__(1024)
final_loss_kernel(const float* __restrict__ tot,
                  const float* __restrict__ pos,
                  const int*   __restrict__ labels,
                  float* __restrict__ out) {
    __shared__ int   hist[64];
    __shared__ float ssum[32];
    __shared__ int   scnt[32];
    const int t = threadIdx.x;
    if (t < 64) hist[t] = 0;
    __syncthreads();
    for (int i = t; i < N_ROWS; i += 1024) atomicAdd(&hist[labels[i]], 1);
    __syncthreads();

    float sum = 0.f;
    int   cnt = 0;
    for (int i = t; i < N_ROWS; i += 1024) {
        int pc = hist[labels[i]] - 1;
        if (pc > 0) {
            float num = pos[i] / (float)pc;
            sum += __logf(tot[i]) - __logf(num);  // -(log num - log den)
            cnt += 1;
        }
    }
#pragma unroll
    for (int m = 16; m >= 1; m >>= 1) {
        sum += __shfl_xor(sum, m, 32);
        cnt += __shfl_xor(cnt, m, 32);
    }
    if ((t & 31) == 0) { ssum[t >> 5] = sum; scnt[t >> 5] = cnt; }
    __syncthreads();
    if (t < 32) {
        sum = ssum[t];
        cnt = scnt[t];
#pragma unroll
        for (int m = 16; m >= 1; m >>= 1) {
            sum += __shfl_xor(sum, m, 32);
            cnt += __shfl_xor(cnt, m, 32);
        }
        if (t == 0) out[0] = sum / (float)cnt;
    }
}

// ---------------------------------------------------------------------------
extern "C" void kernel_launch(void* const* d_in, const int* in_sizes, int n_in,
                              void* d_out, int out_size, void* d_ws, size_t ws_size,
                              hipStream_t stream) {
    const float* emb    = (const float*)d_in[0];
    const int*   labels = (const int*)d_in[1];
    float*       out    = (float*)d_out;

    char* ws = (char*)d_ws;
    __bf16* xhi = (__bf16*)ws;                                       // 16 MB
    __bf16* xlo = (__bf16*)(ws + (size_t)N_ROWS * HDIM * 2);         // 16 MB
    float*  tot = (float*)(ws + (size_t)N_ROWS * HDIM * 4);          // 32 KB
    float*  pos = tot + N_ROWS;                                      // 32 KB

    normalize_split_kernel<<<N_ROWS, 256, 0, stream>>>(emb, xhi, xlo);
    zero_kernel<<<(2 * N_ROWS + 255) / 256, 256, 0, stream>>>(tot, 2 * N_ROWS);

    dim3 grid(N_ROWS / 16, N_ROWS / 128);   // 512 M-tiles x 64 column chunks
    sim_wmma_kernel<<<grid, 32, 0, stream>>>(xhi, xlo, labels, tot, pos);

    final_loss_kernel<<<1, 1024, 0, stream>>>(tot, pos, labels, out);
}
